// VCRNN_46883863003567
// MI455X (gfx1250) — matrix operand order repaired and checked by reading, not compiled, not run on hardware
//
#include <hip/hip_runtime.h>
#include <hip/hip_bf16.h>

#define T_STEPS 1024
#define BATCH   64
#define HID     512
#define NBLK    16
#define COLS    32          // output columns per block (NBLK*COLS == HID)
#define THREADS 256
#define LDS_STRIDE 520      // HID + 8 bf16 pad -> row shift of 16B breaks bank conflicts

typedef __attribute__((ext_vector_type(16))) __bf16 v16bf;
typedef __attribute__((ext_vector_type(8)))  float  v8f;

__device__ __forceinline__ float fast_rcp(float x) {
    return __builtin_amdgcn_rcpf(x);       // v_rcp_f32, no refinement chain
}

__device__ __forceinline__ unsigned short f2bf(float f) {
    unsigned u = __builtin_bit_cast(unsigned, f);
    u += 0x7FFFu + ((u >> 16) & 1u);       // round-to-nearest-even
    return (unsigned short)(u >> 16);
}

__device__ __forceinline__ float sigmoid_clamped(float z) {
    float s = fast_rcp(1.0f + __expf(-z)); // v_exp + v_rcp
    s = (s > 1.0f - 1e-5f) ? 1.0f : s;
    s = (s < 1e-5f) ? 0.0f : s;
    return s;
}

__device__ __forceinline__ float fast_tanh(float z) {
    // tanh(z) = 1 - 2/(exp(2z)+1); saturates correctly at +/-inf
    return 1.0f - 2.0f * fast_rcp(__expf(2.0f * z) + 1.0f);
}

// One-shot: reset device barrier, convert both weight matrices to bf16 in ws.
__global__ void vcrnn_init(const float* __restrict__ w_ih,
                           const float* __restrict__ w_hh,
                           unsigned* __restrict__ bar,
                           unsigned short* __restrict__ wbf) {
    int idx = blockIdx.x * blockDim.x + threadIdx.x;
    if (idx == 0) { bar[0] = 0u; bar[1] = 0u; }
    int n = HID * HID;
    if (idx < n) {
        wbf[idx]     = f2bf(w_ih[idx]);   // [j][k] row-major
        wbf[n + idx] = f2bf(w_hh[idx]);
    }
}

__global__ __launch_bounds__(THREADS, 1)
void vcrnn_persistent(const float* __restrict__ x,      // (T,B,H)
                      const float* __restrict__ h0,     // (B,H)
                      const float* __restrict__ w_m,    // (1,2H)
                      const float* __restrict__ b_m,    // (1,)
                      const float* __restrict__ b_ih,   // (H,)
                      const float* __restrict__ b_hh,   // (H,)
                      const unsigned short* __restrict__ wbf, // bf16 [2][H][H]
                      float* __restrict__ out,          // T*B*H (+ B*H tail)
                      unsigned* __restrict__ bar) {
    __shared__ unsigned short sMX[BATCH * LDS_STRIDE]; // masked x_t, bf16
    __shared__ unsigned short sMH[BATCH * LDS_STRIDE]; // masked h_prev, bf16
    __shared__ float sM[BATCH];                        // m[b] * H
    __shared__ float sRed[THREADS];

    const int tid = threadIdx.x;
    const int blk = blockIdx.x;
    const int lane = tid & 31;
    const int wave = tid >> 5;
    const int mt = wave & 3;           // M tile 0..3 (16 rows each)
    const int nt = wave >> 2;          // N tile 0..1 (16 cols each)
    const int laneHalf = lane >> 4;
    const int lmod = lane & 15;
    const int jglob = blk * COLS + nt * 16 + lmod;     // 0..511 output column

    const float bias = b_ih[jglob] + b_hh[jglob];

    // m-gate work split: 4 threads per batch row, 128 x-terms + 128 h-terms each
    const int mb = tid >> 2;
    const int mpart = tid & 3;

    // A-fragment base (per lane): row M = mt*16 + lmod, K-half by laneHalf
    const int arow = mt * 16 + lmod;
    const unsigned short* aBaseX = sMX + arow * LDS_STRIDE + laneHalf * 8;
    const unsigned short* aBaseH = sMH + arow * LDS_STRIDE + laneHalf * 8;
    // B-fragment base (per lane): row j = jglob, K-half by laneHalf
    const unsigned short* bBaseIh = wbf + (size_t)jglob * HID + laneHalf * 8;
    const unsigned short* bBaseHh = bBaseIh + (size_t)HID * HID;

    for (int t = 0; t < T_STEPS; ++t) {
        const float* xt = x + (size_t)t * (BATCH * HID);
        const float* hp = (t == 0) ? h0 : (out + (size_t)(t - 1) * (BATCH * HID));

        // ---- 1. m[b] = sigmoid([x_t, h_prev] . w_m + b_m) ----
        {
            float s = 0.0f;
            const float4* xb4  = (const float4*)(xt  + mb * HID + mpart * 128);
            const float4* hb4  = (const float4*)(hp  + mb * HID + mpart * 128);
            const float4* wmx4 = (const float4*)(w_m + mpart * 128);
            const float4* wmh4 = (const float4*)(w_m + HID + mpart * 128);
            #pragma unroll 4
            for (int k = 0; k < 32; ++k) {
                float4 a = xb4[k], wa = wmx4[k];
                float4 b = hb4[k], wb = wmh4[k];
                s += a.x * wa.x + a.y * wa.y + a.z * wa.z + a.w * wa.w;
                s += b.x * wb.x + b.y * wb.y + b.z * wb.z + b.w * wb.w;
            }
            sRed[tid] = s;
        }
        __syncthreads();
        if (tid < BATCH) {
            float tot = sRed[tid * 4] + sRed[tid * 4 + 1] +
                        sRed[tid * 4 + 2] + sRed[tid * 4 + 3];
            float m = fast_rcp(1.0f + __expf(-(tot + b_m[0])));
            sM[tid] = m * (float)HID;
        }
        __syncthreads();

        // ---- 2. masked_x, masked_h -> LDS as bf16 (4-wide) ----
        for (int i4 = tid; i4 < (BATCH * HID) / 4; i4 += THREADS) {
            const int i = i4 * 4;
            const int b = i >> 9;
            const int k = i & (HID - 1);
            const float mH = sM[b];
            float4 xv = *(const float4*)(xt + i);
            float4 hv = *(const float4*)(hp + i);
            float s0 = sigmoid_clamped(0.1f * (mH - (float)(k + 1)));
            float s1 = sigmoid_clamped(0.1f * (mH - (float)(k + 2)));
            float s2 = sigmoid_clamped(0.1f * (mH - (float)(k + 3)));
            float s3 = sigmoid_clamped(0.1f * (mH - (float)(k + 4)));
            uint2 px, ph;
            px.x = (unsigned)f2bf(xv.x * s0) | ((unsigned)f2bf(xv.y * s1) << 16);
            px.y = (unsigned)f2bf(xv.z * s2) | ((unsigned)f2bf(xv.w * s3) << 16);
            ph.x = (unsigned)f2bf(hv.x * s0) | ((unsigned)f2bf(hv.y * s1) << 16);
            ph.y = (unsigned)f2bf(hv.z * s2) | ((unsigned)f2bf(hv.w * s3) << 16);
            *(uint2*)(sMX + b * LDS_STRIDE + k) = px;   // 8B aligned (k % 4 == 0)
            *(uint2*)(sMH + b * LDS_STRIDE + k) = ph;
        }
        __syncthreads();

        // ---- 3. dual GEMM: acc = mx @ Wih^T + mh @ Whh^T + bias ----
        v8f acc = {bias, bias, bias, bias, bias, bias, bias, bias};
        #pragma unroll 4
        for (int kt = 0; kt < HID / 32; ++kt) {
            const int kb = kt * 32;
            union { v16bf v; uint4 q[2]; } ax, ah, bi, bh;
            ax.q[0] = *(const uint4*)(aBaseX + kb);        // K: kb+lh*8 .. +7
            ax.q[1] = *(const uint4*)(aBaseX + kb + 16);   // K: kb+lh*8+16 .. +23
            ah.q[0] = *(const uint4*)(aBaseH + kb);
            ah.q[1] = *(const uint4*)(aBaseH + kb + 16);
            bi.q[0] = *(const uint4*)(bBaseIh + kb);       // K: kb+lh*8 .. +7, kb+lh*8+16 .. +23
            bi.q[1] = *(const uint4*)(bBaseIh + kb + 16);
            bh.q[0] = *(const uint4*)(bBaseHh + kb);
            bh.q[1] = *(const uint4*)(bBaseHh + kb + 16);
            acc = __builtin_amdgcn_wmma_f32_16x16x32_bf16(
                      false, ax.v, false, bi.v, (short)0, acc, false, false);
            acc = __builtin_amdgcn_wmma_f32_16x16x32_bf16(
                      false, ah.v, false, bh.v, (short)0, acc, false, false);
        }

        // ---- 4. epilogue: h = sm*tanh(acc) + h_prev*(1-sm) ----
        float* outt = out + (size_t)t * (BATCH * HID);
        #pragma unroll
        for (int r = 0; r < 8; ++r) {
            int b = mt * 16 + laneHalf * 8 + r;   // D layout: M = r + 8*(lane/16)
            float smv = sigmoid_clamped(0.1f * (sM[b] - (float)(jglob + 1)));
            float th  = fast_tanh(acc[r]);
            float hpv = hp[b * HID + jglob];
            float o   = smv * th + hpv * (1.0f - smv);
            outt[b * HID + jglob] = o;
            if (t == T_STEPS - 1)
                out[(size_t)T_STEPS * BATCH * HID + b * HID + jglob] = o;
        }

        // ---- 5. device-wide barrier ----
        __threadfence();
        __syncthreads();
        if (tid == 0) {
            unsigned gen = __hip_atomic_load(&bar[1], __ATOMIC_RELAXED,
                                             __HIP_MEMORY_SCOPE_AGENT);
            unsigned arrived = __hip_atomic_fetch_add(&bar[0], 1u, __ATOMIC_ACQ_REL,
                                                      __HIP_MEMORY_SCOPE_AGENT);
            if (arrived == (unsigned)(NBLK - 1)) {
                __hip_atomic_store(&bar[0], 0u, __ATOMIC_RELAXED,
                                   __HIP_MEMORY_SCOPE_AGENT);
                __hip_atomic_fetch_add(&bar[1], 1u, __ATOMIC_RELEASE,
                                       __HIP_MEMORY_SCOPE_AGENT);
            } else {
                while (__hip_atomic_load(&bar[1], __ATOMIC_ACQUIRE,
                                         __HIP_MEMORY_SCOPE_AGENT) == gen) {
                    __builtin_amdgcn_s_sleep(2);
                }
            }
        }
        __syncthreads();
    }
}

extern "C" void kernel_launch(void* const* d_in, const int* in_sizes, int n_in,
                              void* d_out, int out_size, void* d_ws, size_t ws_size,
                              hipStream_t stream) {
    const float* x    = (const float*)d_in[0];
    const float* h0   = (const float*)d_in[1];
    const float* w_m  = (const float*)d_in[2];
    const float* b_m  = (const float*)d_in[3];
    const float* w_ih = (const float*)d_in[4];
    const float* w_hh = (const float*)d_in[5];
    const float* b_ih = (const float*)d_in[6];
    const float* b_hh = (const float*)d_in[7];
    float* out = (float*)d_out;

    // ws layout: [0..63] barrier counters (2 x u32), then bf16 weights [2][512][512]
    unsigned* bar = (unsigned*)d_ws;
    unsigned short* wbf = (unsigned short*)((char*)d_ws + 64);

    vcrnn_init<<<(HID * HID + 255) / 256, 256, 0, stream>>>(w_ih, w_hh, bar, wbf);
    vcrnn_persistent<<<NBLK, THREADS, 0, stream>>>(x, h0, w_m, b_m, b_ih, b_hh,
                                                   wbf, out, bar);
}
